// Block_24816321036933
// MI455X (gfx1250) — compile-verified, matrix operand order checked by reading, not run on hardware
//
#include <hip/hip_runtime.h>
#include <hip/hip_bf16.h>
#include <math.h>

typedef __bf16 bf16;
typedef __attribute__((ext_vector_type(16))) __bf16 v16bf;
typedef __attribute__((ext_vector_type(8)))  float   v8f;

#define B_      2
#define L_      4096
#define R_      (B_ * L_)      // 8192 token rows
#define DM      768            // d_model
#define DIN     1536           // d_inner
#define DSTATE  64
#define DCONV   4
#define NH      8              // heads
#define HD      192            // headdim
#define DPROJ   3208           // in_proj out width (logical)
#define DPROJP  3264           // in_proj width padded to 64-col tiles
#define DCC     1664           // conv channels
#define DFC     3072           // 4*d_model
#define CHK     64             // ssd chunk
#define NCHUNK  (L_ / CHK)
#define EPS_    1e-5f

// ---------------------------------------------------------------------------
// helpers
// ---------------------------------------------------------------------------
__device__ __forceinline__ float block_reduce256(float v) {
    __shared__ float red[256];
    int t = threadIdx.x;
    red[t] = v;
    __syncthreads();
#pragma unroll
    for (int off = 128; off > 0; off >>= 1) {
        if (t < off) red[t] += red[t + off];
        __syncthreads();
    }
    float r = red[0];
    __syncthreads();
    return r;
}

__device__ __forceinline__ float silu_f(float x) { return x / (1.0f + expf(-x)); }

// Load a 16x32 bf16 WMMA operand fragment for one lane.
// p points at (row base + half*8 + k0); elements 0..7 -> K offsets 0..7,
// elements 8..15 -> K offsets 16..23 (ISA 16-bit A/B layout, wave32).
__device__ __forceinline__ v16bf load_frag16(const bf16* __restrict__ p) {
    v16bf f;
#pragma unroll
    for (int i = 0; i < 8; i++) { f[i] = p[i]; f[i + 8] = p[i + 16]; }
    return f;
}

// ---------------------------------------------------------------------------
// fp32 -> bf16 cast with zero tail padding (pads weight rows to tile multiple)
// ---------------------------------------------------------------------------
__global__ void cast_pad_bf16_kernel(const float* __restrict__ in, bf16* __restrict__ out,
                                     size_t nvalid, size_t ntotal) {
    size_t i = (size_t)blockIdx.x * blockDim.x + threadIdx.x;
    if (i < ntotal) out[i] = (i < nvalid) ? (bf16)in[i] : (bf16)0.0f;
}

// ---------------------------------------------------------------------------
// WMMA GEMM: out[M,Nld] = A[M,K](bf16) @ W[Nld,K]^T(bf16), fp32 accum.
// Nld padded to a multiple of 64 -> no guards. K/32 must be even (768/1536/
// 3072 all qualify). block = 256 threads (8 waves); each wave computes a
// 16x64 strip. K-loop unrolled x2 with ping-pong fragment buffers so the
// software pipeline needs no rotation moves.
// EPI=0: store fp32.  EPI=1: exact GELU then store bf16.
// ---------------------------------------------------------------------------
template <int EPI>
__global__ __launch_bounds__(256) void gemm_bf16_wmma(
    const bf16* __restrict__ A, const bf16* __restrict__ W,
    void* __restrict__ outp, int M, int Nld, int K) {
    const int wave = threadIdx.x >> 5;
    const int lane = threadIdx.x & 31;
    const int half = lane >> 4;
    const int lid  = lane & 15;
    const int m0   = blockIdx.x * 128 + wave * 16;
    const int mrow = m0 + lid;                 // A row this lane supplies
    const int nbase = blockIdx.y * 64;

    v8f acc[4] = {};

    const bf16* ap = A + (size_t)mrow * K + half * 8;
    const bf16* wp[4];
#pragma unroll
    for (int t = 0; t < 4; t++)
        wp[t] = W + (size_t)(nbase + t * 16 + lid) * K + half * 8;

    // prologue: fragment set 0 holds K-step 0
    v16bf a0 = load_frag16(ap);
    v16bf b0[4];
#pragma unroll
    for (int t = 0; t < 4; t++) b0[t] = load_frag16(wp[t]);

    v16bf a1;
    v16bf b1[4];

    // main loop, 2 K-steps per iteration, ping-pong buffers (no rotation movs).
    // invariant at loop top: buffer 0 holds fragments for step (k0-32).
    int k0 = 32;
    for (; k0 < K - 32; k0 += 64) {
        a1 = load_frag16(ap + k0);
#pragma unroll
        for (int t = 0; t < 4; t++) b1[t] = load_frag16(wp[t] + k0);
#pragma unroll
        for (int t = 0; t < 4; t++)
            acc[t] = __builtin_amdgcn_wmma_f32_16x16x32_bf16(
                false, a0, false, b0[t], (short)0, acc[t], false, false);
        a0 = load_frag16(ap + k0 + 32);
#pragma unroll
        for (int t = 0; t < 4; t++) b0[t] = load_frag16(wp[t] + k0 + 32);
#pragma unroll
        for (int t = 0; t < 4; t++)
            acc[t] = __builtin_amdgcn_wmma_f32_16x16x32_bf16(
                false, a1, false, b1[t], (short)0, acc[t], false, false);
    }
    // epilogue: k0 == K-32 here; buffer 0 holds step K-64.
    a1 = load_frag16(ap + k0);
#pragma unroll
    for (int t = 0; t < 4; t++) b1[t] = load_frag16(wp[t] + k0);
#pragma unroll
    for (int t = 0; t < 4; t++)
        acc[t] = __builtin_amdgcn_wmma_f32_16x16x32_bf16(
            false, a0, false, b0[t], (short)0, acc[t], false, false);
#pragma unroll
    for (int t = 0; t < 4; t++)
        acc[t] = __builtin_amdgcn_wmma_f32_16x16x32_bf16(
            false, a1, false, b1[t], (short)0, acc[t], false, false);

    // store (C/D layout: VGPR j -> row m0 + j + 8*half, col nbase + t*16 + lid)
#pragma unroll
    for (int t = 0; t < 4; t++) {
        int n = nbase + t * 16 + lid;
#pragma unroll
        for (int j = 0; j < 8; j++) {
            int mr = m0 + j + 8 * half;
            float v = acc[t][j];
            if (EPI == 0) {
                ((float*)outp)[(size_t)mr * Nld + n] = v;
            } else {
                float g = 0.5f * v * (1.0f + erff(v * 0.70710678118654752f));
                ((bf16*)outp)[(size_t)mr * Nld + n] = (bf16)g;
            }
        }
    }
}

// ---------------------------------------------------------------------------
// RMSNorm over 768 cols -> bf16
// ---------------------------------------------------------------------------
__global__ __launch_bounds__(256) void rmsnorm768_bf16(
    const float* __restrict__ x, const float* __restrict__ w, bf16* __restrict__ out) {
    int row = blockIdx.x;
    const float* xr = x + (size_t)row * DM;
    float s[3]; float ss = 0.0f;
#pragma unroll
    for (int j = 0; j < 3; j++) {
        float v = xr[threadIdx.x + j * 256];
        s[j] = v; ss += v * v;
    }
    float tot = block_reduce256(ss);
    float r = rsqrtf(tot / (float)DM + EPS_);
    bf16* o = out + (size_t)row * DM;
#pragma unroll
    for (int j = 0; j < 3; j++) {
        int i = threadIdx.x + j * 256;
        o[i] = (bf16)(s[j] * r * w[i]);
    }
}

// ---------------------------------------------------------------------------
// Depthwise causal conv (width 4) + SiLU over xBC channels of zxbcdt (padded)
// ---------------------------------------------------------------------------
__global__ void conv_silu_kernel(const float* __restrict__ zx, const float* __restrict__ cw,
                                 const float* __restrict__ cb, float* __restrict__ out) {
    size_t idx = (size_t)blockIdx.x * blockDim.x + threadIdx.x;
    if (idx >= (size_t)R_ * DCC) return;
    int c = (int)(idx % DCC);
    int t = (int)(idx / DCC);   // b*L + l
    int l = t % L_;
    float acc = cb[c];
#pragma unroll
    for (int i = 0; i < DCONV; i++) {
        int ls = l - (DCONV - 1) + i;
        if (ls >= 0)
            acc += zx[(size_t)(t - l + ls) * DPROJP + DIN + c] * cw[c * DCONV + i];
    }
    out[idx] = silu_f(acc);
}

// ---------------------------------------------------------------------------
// dt = softplus(dt_raw + dt_bias)
// ---------------------------------------------------------------------------
__global__ void dt_softplus_kernel(const float* __restrict__ zx, const float* __restrict__ db,
                                   float* __restrict__ out) {
    size_t idx = (size_t)blockIdx.x * blockDim.x + threadIdx.x;
    if (idx >= (size_t)R_ * NH) return;
    int h = (int)(idx % NH);
    size_t t = idx / NH;
    float v = zx[t * DPROJP + DIN + DCC + h] + db[h];
    out[idx] = (v > 20.0f) ? v : log1pf(expf(v));
}

// ---------------------------------------------------------------------------
// SSD chunked selective scan. One workgroup per (batch, head), 192 threads =
// one per head-dim lane p. State S[p][64] in registers; chunk B/C/decay in LDS.
// ---------------------------------------------------------------------------
__global__ __launch_bounds__(192) void ssd_kernel(
    const float* __restrict__ xc,     // conv_out [R_, DCC]: xh | B | C
    const float* __restrict__ dtsp,   // [R_, NH]
    const float* __restrict__ A_log, const float* __restrict__ Dv,
    float* __restrict__ y) {          // [R_, DIN]
    const int b = blockIdx.x >> 3;
    const int h = blockIdx.x & 7;
    const int p = threadIdx.x;        // 0..191

    __shared__ float sB[CHK][DSTATE + 1];
    __shared__ float sC[CHK][DSTATE + 1];
    __shared__ float sG[CHK][CHK + 1];
    __shared__ float sDt[CHK], sAcs[CHK], sW[CHK];

    float S[DSTATE];
#pragma unroll
    for (int n = 0; n < DSTATE; n++) S[n] = 0.0f;

    const float A_h = -expf(A_log[h]);
    const float D_h = Dv[h];
    const size_t rowb = (size_t)b * L_;
    const size_t xoff = (size_t)h * HD + p;

    for (int c = 0; c < NCHUNK; c++) {
        const int l0 = c * CHK;
        // stage B/C/dt for this chunk
        for (int i = p; i < CHK * DSTATE; i += 192) {
            int k = i >> 6, n = i & 63;
            size_t r = (rowb + l0 + k) * DCC;
            sB[k][n] = xc[r + DIN + n];
            sC[k][n] = xc[r + DIN + DSTATE + n];
        }
        if (p < CHK) sDt[p] = dtsp[(rowb + l0 + p) * NH + h];
        __syncthreads();
        if (p == 0) {                       // cumsum of A_h*dt within chunk
            float a = 0.0f;
            for (int k = 0; k < CHK; k++) { a += A_h * sDt[k]; sAcs[k] = a; }
        }
        __syncthreads();
        const float AcsL = sAcs[CHK - 1];
        // G[k][s] = (C_k . B_s) * exp(Acs_k - Acs_s) * dt_s  (lower triangular)
        for (int i = p; i < CHK * CHK; i += 192) {
            int k = i >> 6, s = i & 63;
            float g = 0.0f;
            if (s <= k) {
                float dot = 0.0f;
                for (int n = 0; n < DSTATE; n++) dot += sC[k][n] * sB[s][n];
                g = dot * expf(sAcs[k] - sAcs[s]) * sDt[s];
            }
            sG[k][s] = g;
        }
        if (p < CHK) sW[p] = sDt[p] * expf(AcsL - sAcs[p]);
        __syncthreads();

        // diagonal block: yacc[k] = sum_s G[k][s] * xh[s][p]
        float yacc[CHK];
#pragma unroll
        for (int k = 0; k < CHK; k++) yacc[k] = 0.0f;
        for (int s = 0; s < CHK; s++) {
            float xs = xc[(rowb + l0 + s) * DCC + xoff];
#pragma unroll
            for (int k = 0; k < CHK; k++) yacc[k] += sG[k][s] * xs;
        }

        // off-diagonal via carried state + D skip, write y
#pragma unroll
        for (int k = 0; k < CHK; k++) {
            float yo = 0.0f;
#pragma unroll
            for (int n = 0; n < DSTATE; n++) yo += sC[k][n] * S[n];
            float xk = xc[(rowb + l0 + k) * DCC + xoff];
            y[(rowb + l0 + k) * DIN + xoff] = yacc[k] + yo * expf(sAcs[k]) + xk * D_h;
        }

        // state recurrence: S = exp(chunk_sum)*S + sum_s B_s * w_s * xh_s
        const float eA = expf(AcsL);
#pragma unroll
        for (int n = 0; n < DSTATE; n++) S[n] *= eA;
        for (int s = 0; s < CHK; s++) {
            float ws = sW[s] * xc[(rowb + l0 + s) * DCC + xoff];
#pragma unroll
            for (int n = 0; n < DSTATE; n++) S[n] += sB[s][n] * ws;
        }
        __syncthreads();
    }
}

// ---------------------------------------------------------------------------
// gated RMSNorm: out = rmsnorm(y * silu(z), w) -> bf16   (z read from zxbcdt)
// ---------------------------------------------------------------------------
__global__ __launch_bounds__(256) void gated_rmsnorm_bf16(
    const float* __restrict__ y, const float* __restrict__ zx,
    const float* __restrict__ w, bf16* __restrict__ out) {
    int row = blockIdx.x;
    const float* yr = y + (size_t)row * DIN;
    const float* zr = zx + (size_t)row * DPROJP;
    float v[6]; float ss = 0.0f;
#pragma unroll
    for (int j = 0; j < 6; j++) {
        int i = threadIdx.x + j * 256;
        float g = yr[i] * silu_f(zr[i]);
        v[j] = g; ss += g * g;
    }
    float tot = block_reduce256(ss);
    float r = rsqrtf(tot / (float)DIN + EPS_);
    bf16* o = out + (size_t)row * DIN;
#pragma unroll
    for (int j = 0; j < 6; j++) {
        int i = threadIdx.x + j * 256;
        o[i] = (bf16)(v[j] * r * w[i]);
    }
}

// ---------------------------------------------------------------------------
// x1 = x + g (fp32) ; h = rmsnorm(x1, w) -> bf16
// ---------------------------------------------------------------------------
__global__ __launch_bounds__(256) void resid_rmsnorm_bf16(
    const float* __restrict__ x, const float* __restrict__ g, float* __restrict__ x1,
    const float* __restrict__ w, bf16* __restrict__ hout) {
    int row = blockIdx.x;
    size_t base = (size_t)row * DM;
    float s[3]; float ss = 0.0f;
#pragma unroll
    for (int j = 0; j < 3; j++) {
        int i = threadIdx.x + j * 256;
        float v = x[base + i] + g[base + i];
        x1[base + i] = v; s[j] = v; ss += v * v;
    }
    float tot = block_reduce256(ss);
    float r = rsqrtf(tot / (float)DM + EPS_);
#pragma unroll
    for (int j = 0; j < 3; j++) {
        int i = threadIdx.x + j * 256;
        hout[base + i] = (bf16)(s[j] * r * w[i]);
    }
}

__global__ void add_kernel(const float* __restrict__ a, const float* __restrict__ b,
                           float* __restrict__ o, size_t n) {
    size_t i = (size_t)blockIdx.x * blockDim.x + threadIdx.x;
    if (i < n) o[i] = a[i] + b[i];
}

// ---------------------------------------------------------------------------
// launch
// ---------------------------------------------------------------------------
extern "C" void kernel_launch(void* const* d_in, const int* in_sizes, int n_in,
                              void* d_out, int out_size, void* d_ws, size_t ws_size,
                              hipStream_t stream) {
    const float* x      = (const float*)d_in[0];
    const float* wn1    = (const float*)d_in[1];
    const float* wn2    = (const float*)d_in[2];
    const float* W_in   = (const float*)d_in[3];
    const float* conv_w = (const float*)d_in[4];
    const float* conv_b = (const float*)d_in[5];
    const float* dt_b   = (const float*)d_in[6];
    const float* A_log  = (const float*)d_in[7];
    const float* Dv     = (const float*)d_in[8];
    const float* wmn    = (const float*)d_in[9];
    const float* W_out  = (const float*)d_in[10];
    const float* W_fc   = (const float*)d_in[11];
    const float* W_pr   = (const float*)d_in[12];
    (void)in_sizes; (void)n_in; (void)out_size; (void)ws_size;

    char* wsp = (char*)d_ws;
    auto carve = [&](size_t bytes) -> void* {
        void* p = (void*)wsp;
        wsp += (bytes + 255) & ~(size_t)255;
        return p;
    };
    float* zx   = (float*)carve((size_t)R_ * DPROJP * 4);  // in_proj output (padded)
    float* cv   = (float*)carve((size_t)R_ * DCC * 4);     // conv+silu output
    float* dts  = (float*)carve((size_t)R_ * NH * 4);      // softplus dt
    float* yb   = (float*)carve((size_t)R_ * DIN * 4);     // ssd output
    float* x1   = (float*)carve((size_t)R_ * DM * 4);      // first residual
    float* go   = (float*)carve((size_t)R_ * DM * 4);      // gemm fp32 out (reused)
    bf16*  actA = (bf16*)carve((size_t)R_ * DFC * 2);      // wide bf16 acts (reused)
    bf16*  actB = (bf16*)carve((size_t)R_ * DM * 2);       // narrow bf16 acts (reused)
    bf16*  wbuf = (bf16*)carve((size_t)DPROJP * DM * 2);   // bf16 weight staging (padded)

    // 1) u = rmsnorm(x, w_norm1) -> bf16
    rmsnorm768_bf16<<<R_, 256, 0, stream>>>(x, wn1, actB);
    // 2) zxbcdt = u @ W_in^T  (weight padded 3208 -> 3264 rows)
    { size_t nv = (size_t)DPROJ * DM, nt = (size_t)DPROJP * DM;
      cast_pad_bf16_kernel<<<(unsigned)((nt + 255) / 256), 256, 0, stream>>>(W_in, wbuf, nv, nt); }
    gemm_bf16_wmma<0><<<dim3(R_ / 128, DPROJP / 64), 256, 0, stream>>>(
        actB, wbuf, (void*)zx, R_, DPROJP, DM);
    // 3) depthwise conv + silu ; softplus dt
    { size_t n = (size_t)R_ * DCC;
      conv_silu_kernel<<<(unsigned)((n + 255) / 256), 256, 0, stream>>>(zx, conv_w, conv_b, cv); }
    { size_t n = (size_t)R_ * NH;
      dt_softplus_kernel<<<(unsigned)((n + 255) / 256), 256, 0, stream>>>(zx, dt_b, dts); }
    // 4) SSD scan
    ssd_kernel<<<B_ * NH, 192, 0, stream>>>(cv, dts, A_log, Dv, yb);
    // 5) gated rmsnorm -> bf16
    gated_rmsnorm_bf16<<<R_, 256, 0, stream>>>(yb, zx, wmn, actA);
    // 6) out_proj
    { size_t n = (size_t)DM * DIN;
      cast_pad_bf16_kernel<<<(unsigned)((n + 255) / 256), 256, 0, stream>>>(W_out, wbuf, n, n); }
    gemm_bf16_wmma<0><<<dim3(R_ / 128, DM / 64), 256, 0, stream>>>(
        actA, wbuf, (void*)go, R_, DM, DIN);
    // 7) residual + rmsnorm -> bf16
    resid_rmsnorm_bf16<<<R_, 256, 0, stream>>>(x, go, x1, wn2, actB);
    // 8) fc with fused exact-GELU epilogue -> bf16
    { size_t n = (size_t)DFC * DM;
      cast_pad_bf16_kernel<<<(unsigned)((n + 255) / 256), 256, 0, stream>>>(W_fc, wbuf, n, n); }
    gemm_bf16_wmma<1><<<dim3(R_ / 128, DFC / 64), 256, 0, stream>>>(
        actB, wbuf, (void*)actA, R_, DFC, DM);
    // 9) proj
    { size_t n = (size_t)DM * DFC;
      cast_pad_bf16_kernel<<<(unsigned)((n + 255) / 256), 256, 0, stream>>>(W_pr, wbuf, n, n); }
    gemm_bf16_wmma<0><<<dim3(R_ / 128, DM / 64), 256, 0, stream>>>(
        actA, wbuf, (void*)go, R_, DM, DFC);
    // 10) final residual
    { size_t n = (size_t)R_ * DM;
      add_kernel<<<(unsigned)((n + 255) / 256), 256, 0, stream>>>(x1, go, (float*)d_out, n); }
}
